// RNNSoftPlus_9285719294337
// MI455X (gfx1250) — matrix operand order repaired and checked by reading, not compiled, not run on hardware
//
#include <hip/hip_runtime.h>

typedef __attribute__((ext_vector_type(16))) __bf16 v16bf;
typedef __attribute__((ext_vector_type(8)))  float  v8f;
typedef unsigned short ushort_t;

struct alignas(16) U128 { unsigned int a, b, c, d; };
union FragU { v16bf v; U128 q[2]; };

__device__ __forceinline__ unsigned short f2bf(float f) {
    unsigned int u = __float_as_uint(f);
    u += 0x7FFFu + ((u >> 16) & 1u);          // round-to-nearest-even
    return (unsigned short)(u >> 16);
}

// A-fragment (16x32 bf16, M x K), row-major source, ld = 1024.
// Lane l: row = l&15; chunk0 = K[kk + ((l>=16)?16:0) .. +8), chunk1 = K[kk+8+((l>=16)?16:0) .. +8)
__device__ __forceinline__ v16bf load_a_frag(const ushort_t* __restrict__ row_base, int kk, int lane) {
    FragU f;
    const int kh = (lane >> 4) << 4;   // 0 or 16
    f.q[0] = *reinterpret_cast<const U128*>(row_base + kk + kh);
    f.q[1] = *reinterpret_cast<const U128*>(row_base + kk + 8 + kh);
    return f.v;
}

// B-fragment (32x16 bf16, K x N) from W[N,K] row-major: column n of B = row n of W.
// p points at 16 contiguous K values for this lane's (row, K-range).
__device__ __forceinline__ v16bf load_b_frag(const ushort_t* __restrict__ p) {
    FragU f;
    f.q[0] = *reinterpret_cast<const U128*>(p);
    f.q[1] = *reinterpret_cast<const U128*>(p + 8);
    return f.v;
}

__device__ __forceinline__ v8f wmma_bf16(v16bf a, v16bf b, v8f c) {
    return __builtin_amdgcn_wmma_f32_16x16x32_bf16(false, a, false, b, (short)0, c, false, false);
}

// ---------------- elementwise prep kernels ----------------

__global__ void f32_to_bf16_kernel(const float* __restrict__ src, ushort_t* __restrict__ dst, int n) {
    for (int i = blockIdx.x * blockDim.x + threadIdx.x; i < n; i += gridDim.x * blockDim.x)
        dst[i] = f2bf(src[i]);
}

// hidden: [2][128][1024] fp32 -> hbuf[dir][phase0][128*1024] bf16 (dir stride = 2*131072)
__global__ void init_h_kernel(const float* __restrict__ hidden, ushort_t* __restrict__ hbuf) {
    int i = blockIdx.x * blockDim.x + threadIdx.x;
    if (i < 262144) {
        int dir = i >> 17;
        int off = i & 131071;
        hbuf[(size_t)dir * 262144 + off] = f2bf(hidden[i]);
    }
}

// ---------------- pre = x . Wi^T + bi  (M=32768, N=1024, K=1024) ----------------
// Block: 8 waves; wave w computes rows [blockIdx.y*128 + 16w, +16), cols [blockIdx.x*64, +64).
// B tile (64 cols x 64 K) staged cooperatively into LDS with async-to-LDS DMA,
// double buffered and overlapped with WMMA; A rows streamed from global.
// One barrier per K-chunk: reaching it proves (a) every wave's async loads for the
// current chunk completed (each waited its own ASYNCcnt first) and (b) every wave's
// DScnt-tracked reads of the buffer about to be overwritten have retired.

__global__ __launch_bounds__(256) void gemm_pre_kernel(
    const ushort_t* __restrict__ xb, const ushort_t* __restrict__ Wib,
    const float* __restrict__ bi, float* __restrict__ pre)
{
    __shared__ alignas(16) ushort_t lb[2][64][64];   // 2 x 8 KB B-tile buffers

    const int tid  = threadIdx.x;
    const int lane = tid & 31;
    const int w    = tid >> 5;
    const int rowTile = blockIdx.y * 128 + w * 16;
    const int colTile = blockIdx.x * 64;

    // --- async-copy indexing: 512 x 16B transfers per chunk, 2 per thread ---
    const int r0 = tid >> 3;                  // rows 0..31
    const int r1 = r0 + 32;                   // rows 32..63
    const int kc = (tid & 7) * 8;             // K sub-offset (elements)
    const unsigned long long g0 =
        (unsigned long long)(const void*)(Wib + (size_t)(colTile + r0) * 1024 + kc);
    const unsigned long long g1 =
        (unsigned long long)(const void*)(Wib + (size_t)(colTile + r1) * 1024 + kc);
    unsigned lds0[2], lds1[2];
    lds0[0] = (unsigned)(unsigned long long)(void*)&lb[0][r0][kc];
    lds0[1] = (unsigned)(unsigned long long)(void*)&lb[1][r0][kc];
    lds1[0] = (unsigned)(unsigned long long)(void*)&lb[0][r1][kc];
    lds1[1] = (unsigned)(unsigned long long)(void*)&lb[1][r1][kc];

    auto issue = [&](int chunk, int buf) {
        const unsigned long long ga0 = g0 + (unsigned long long)chunk * 128ull; // 64 elems * 2B
        const unsigned long long ga1 = g1 + (unsigned long long)chunk * 128ull;
        asm volatile("global_load_async_to_lds_b128 %0, %1, off"
                     :: "v"(lds0[buf]), "v"(ga0) : "memory");
        asm volatile("global_load_async_to_lds_b128 %0, %1, off"
                     :: "v"(lds1[buf]), "v"(ga1) : "memory");
    };

    const ushort_t* abase = xb + (size_t)(rowTile + (lane & 15)) * 1024;
    const int ko = (lane >> 4) << 4;   // 0 or 16 (B-frag K range select)
    const int nl = lane & 15;

    v8f acc0 = {}, acc1 = {}, acc2 = {}, acc3 = {};

    issue(0, 0);
    for (int c = 0; c < 16; ++c) {           // 16 K-chunks of 64
        asm volatile("s_wait_asynccnt 0x0" ::: "memory");
        __syncthreads();                      // chunk c resident; prior buffer reads retired
        if (c + 1 < 16) issue(c + 1, (c + 1) & 1);

        const int kb = c & 1;
        #pragma unroll
        for (int s = 0; s < 2; ++s) {         // two K=32 WMMA steps per chunk
            const int kl = s * 32;
            v16bf a = load_a_frag(abase, c * 64 + kl, lane);
            acc0 = wmma_bf16(a, load_b_frag(&lb[kb][ 0 + nl][kl + ko]), acc0);
            acc1 = wmma_bf16(a, load_b_frag(&lb[kb][16 + nl][kl + ko]), acc1);
            acc2 = wmma_bf16(a, load_b_frag(&lb[kb][32 + nl][kl + ko]), acc2);
            acc3 = wmma_bf16(a, load_b_frag(&lb[kb][48 + nl][kl + ko]), acc3);
        }
    }

    const int mo = (lane >> 4) << 3;   // 0 or 8
    const float b0 = bi[colTile + nl], b1 = bi[colTile + 16 + nl];
    const float b2 = bi[colTile + 32 + nl], b3 = bi[colTile + 48 + nl];
    #pragma unroll
    for (int i = 0; i < 8; ++i) {
        float* p = pre + (size_t)(rowTile + mo + i) * 1024 + colTile + nl;
        p[0]  = acc0[i] + b0;
        p[16] = acc1[i] + b1;
        p[32] = acc2[i] + b2;
        p[48] = acc3[i] + b3;
    }
}

// ---------------- one recurrent step, both directions ----------------
// h_next = softplus5(pre[t] + h . Wh^T + bh); grid (16 Ntiles, 8 Mtiles, 2 dirs).
// 8 waves split K into 4 slices x 2 column halves; deterministic LDS reduction.

__global__ __launch_bounds__(256) void rnn_step_kernel(
    const ushort_t* __restrict__ hprev_f, ushort_t* __restrict__ hnext_f,
    const ushort_t* __restrict__ hprev_b, ushort_t* __restrict__ hnext_b,
    const float* __restrict__ pre, const ushort_t* __restrict__ Whb,
    const float* __restrict__ bh, float* __restrict__ out,
    int tf_pre, int tb_pre, int tf_out, int tb_out)
{
    __shared__ float lpart[4][16][64];   // 16 KB

    const int dir = blockIdx.z;
    const ushort_t* hprev = dir ? hprev_b : hprev_f;
    ushort_t*       hnext = dir ? hnext_b : hnext_f;
    const int tpre = dir ? tb_pre : tf_pre;
    const int tout = dir ? tb_out : tf_out;

    const int lane = threadIdx.x & 31;
    const int w    = threadIdx.x >> 5;
    const int ksl  = w >> 1;           // K slice: [ksl*256, +256)
    const int ch   = w & 1;            // column half: 32 cols
    const int rowTile = blockIdx.y * 16;
    const int colTile = blockIdx.x * 64;

    const ushort_t* abase = hprev + (size_t)(rowTile + (lane & 15)) * 1024;
    const int ko = (lane >> 4) << 4;
    const ushort_t* wb0 = Whb + (size_t)(colTile + ch * 32 +  0 + (lane & 15)) * 1024 + ko;
    const ushort_t* wb1 = Whb + (size_t)(colTile + ch * 32 + 16 + (lane & 15)) * 1024 + ko;

    v8f acc0 = {}, acc1 = {};
    const int k0 = ksl * 256;
    #pragma unroll 4
    for (int kk = k0; kk < k0 + 256; kk += 32) {
        v16bf a = load_a_frag(abase, kk, lane);
        acc0 = wmma_bf16(a, load_b_frag(wb0 + kk), acc0);
        acc1 = wmma_bf16(a, load_b_frag(wb1 + kk), acc1);
    }

    const int mo = (lane >> 4) << 3;
    const int nl = lane & 15;
    #pragma unroll
    for (int i = 0; i < 8; ++i) {
        lpart[ksl][mo + i][ch * 32 + nl]      = acc0[i];
        lpart[ksl][mo + i][ch * 32 + 16 + nl] = acc1[i];
    }
    __syncthreads();

    // deterministic cross-wave reduction + fused softplus5 + scattered stores
    #pragma unroll
    for (int e = 0; e < 4; ++e) {
        const int idx = threadIdx.x + e * 256;   // 0..1023 over 16x64 tile
        const int r = idx >> 6, c = idx & 63;
        const int row = rowTile + r;
        const int col = colTile + c;
        float y = ((lpart[0][r][c] + lpart[1][r][c]) + (lpart[2][r][c] + lpart[3][r][c]));
        y += pre[((size_t)row * 256 + tpre) * 1024 + col] + bh[col];
        const float z = 5.0f * y;
        const float res = (z > 20.0f) ? y : 0.2f * __logf(1.0f + __expf(z));
        out[((size_t)row * 256 + tout) * 2048 + (size_t)dir * 1024 + col] = res;
        hnext[(size_t)row * 1024 + col] = f2bf(res);
    }
}

// ---------------- host launcher ----------------

extern "C" void kernel_launch(void* const* d_in, const int* in_sizes, int n_in,
                              void* d_out, int out_size, void* d_ws, size_t ws_size,
                              hipStream_t stream)
{
    (void)in_sizes; (void)n_in; (void)out_size; (void)ws_size;
    const float* x      = (const float*)d_in[0];   // [128,256,1024]
    const float* hidden = (const float*)d_in[1];   // [2,128,1024]
    const float* Wi     = (const float*)d_in[2];   // [1024,1024]
    const float* bi     = (const float*)d_in[3];   // [1024]
    const float* Wh     = (const float*)d_in[4];   // [1024,1024]
    const float* bh     = (const float*)d_in[5];   // [1024]
    float* out = (float*)d_out;                    // [128,256,2048]

    char* ws = (char*)d_ws;
    ushort_t* xb   = (ushort_t*)(ws);              // 33,554,432 bf16 = 64 MiB
    ushort_t* Wib  = (ushort_t*)(ws + 67108864);   // 1,048,576 bf16 = 2 MiB
    ushort_t* Whb  = (ushort_t*)(ws + 69206016);   // 2 MiB
    ushort_t* hbuf = (ushort_t*)(ws + 71303168);   // 2 dirs x 2 phases x 131072 bf16 = 1 MiB
    float*    pre  = (float*)   (ws + 72351744);   // 33,554,432 f32 = 128 MiB

    // prep: bf16 conversions + initial hidden state
    f32_to_bf16_kernel<<<4096, 256, 0, stream>>>(x,  xb,  33554432);
    f32_to_bf16_kernel<<<1024, 256, 0, stream>>>(Wi, Wib, 1048576);
    f32_to_bf16_kernel<<<1024, 256, 0, stream>>>(Wh, Whb, 1048576);
    init_h_kernel<<<1024, 256, 0, stream>>>(hidden, hbuf);

    // pre = x . Wi^T + bi  (layout [B][T][H] since M = b*256 + t)
    gemm_pre_kernel<<<dim3(16, 256), 256, 0, stream>>>(xb, Wib, bi, pre);

    // 256 recurrent steps; fwd + bwd concurrently; ping-pong h buffers
    const int HB = 131072;
    for (int j = 0; j < 256; ++j) {
        const int ph = j & 1;
        const ushort_t* hpf = hbuf + ph * HB;
        ushort_t*       hnf = hbuf + (ph ^ 1) * HB;
        const ushort_t* hpb = hbuf + 2 * HB + ph * HB;
        ushort_t*       hnb = hbuf + 2 * HB + (ph ^ 1) * HB;
        rnn_step_kernel<<<dim3(16, 8, 2), 256, 0, stream>>>(
            hpf, hnf, hpb, hnb, pre, Whb, bh, out,
            /*tf_pre=*/j, /*tb_pre=*/255 - j,
            /*tf_out=*/(239 - j) & 255, /*tb_out=*/j);
    }
}
